// MoELinear_79620103733347
// MI455X (gfx1250) — matrix operand order repaired and checked by reading, not compiled, not run on hardware
//
#include <hip/hip_runtime.h>
#include <hip/hip_bf16.h>
#include <stdint.h>

// Problem constants (from the reference)
#define N_LORAS 28
#define RANK    16
#define RMOE    (N_LORAS * RANK)   // 448
#define N_EXP   (N_LORAS + 1)      // 29 gate rows
#define MTOK    (4 * 2048)         // 8192 tokens
#define D_IN    4096
#define D_OUT   4096
#define SCALING 2.0f               // LORA_ALPHA / RANK

// GEMM tiling: 256x128 block tile, 8 waves in 4(M) x 2(N) grid,
// 64x64 per-wave tile = 4x4 WMMA tiles -> 16 ds_load_b128 : 16 wmma per K-step.
#define BM      256
#define BN      128
#define KB      32                 // K per WMMA step (bf16 16x16x32)
#define LDSROW  80                 // 32 bf16 = 64B + 16B pad -> conflict-free b128 reads

typedef __attribute__((ext_vector_type(16))) __bf16 v16bf;
typedef __attribute__((ext_vector_type(8)))  float  v8f;

union FragU { uint4 q[2]; v16bf v; };

// ---------------------------------------------------------------- helpers ---

#if __has_builtin(__builtin_amdgcn_cvt_pk_bf16_f32)
typedef __attribute__((ext_vector_type(2))) __bf16 v2bf;
__device__ __forceinline__ unsigned int cvt2_bf16(float lo, float hi) {
  union { v2bf v; unsigned int u; } c;
  c.v = __builtin_amdgcn_cvt_pk_bf16_f32(lo, hi);
  return c.u;
}
#else
// round-half-up to bf16, pack two with one v_perm_b32: 3 VALU per 2 floats
__device__ __forceinline__ unsigned int cvt2_bf16(float lo, float hi) {
  unsigned int a = __float_as_uint(lo) + 0x8000u;
  unsigned int b = __float_as_uint(hi) + 0x8000u;
  // result bytes: {b[3], b[2], a[3], a[2]}  (sel indexes: 0-3 -> 2nd op, 4-7 -> 1st)
  return __builtin_amdgcn_perm(b, a, 0x07060302u);
}
#endif

__device__ __forceinline__ uint2 pack4_bf16(float4 v) {
  uint2 r;
  r.x = cvt2_bf16(v.x, v.y);
  r.y = cvt2_bf16(v.z, v.w);
  return r;
}

// A-fragment (16x32 bf16): lanes 0-15 hold K0..7 (v0..3) and K16..23 (v4..7);
// lanes 16-31 hold K8..15 and K24..31.  (ISA 7.12.2, 16-bit A 16x32)
__device__ __forceinline__ v16bf load_frag_a(const unsigned char* buf, int row, int hsel) {
  FragU f;
  const unsigned char* p = buf + row * LDSROW + hsel * 16;
  f.q[0] = *reinterpret_cast<const uint4*>(p);
  f.q[1] = *reinterpret_cast<const uint4*>(p + 32);
  return f.v;
}

// B-fragment (32x16 bf16): lane = column N; lanes 0-15 hold K0..7 and K16..23,
// lanes 16-31 hold K8..15 and K24..31 (same K order as the A-fragment).
__device__ __forceinline__ v16bf load_frag_b(const unsigned char* buf, int row, int hsel) {
  FragU f;
  const unsigned char* p = buf + row * LDSROW + hsel * 16;
  f.q[0] = *reinterpret_cast<const uint4*>(p);
  f.q[1] = *reinterpret_cast<const uint4*>(p + 32);
  return f.v;
}

// Staging register block: A 256x32 (8 chunks) + B 128x32 (4 chunks) per thread.
struct StageRegs { float4 a[8]; float4 b[4]; };

// Issue the 12 global_load_b128 for one K-step; no wait, results land later.
__device__ __forceinline__ void stage_load(StageRegs& r,
                                           const float* __restrict__ A, int lda,
                                           const float* __restrict__ Bw, int ldb,
                                           int rowA0, int rowB0, int kbase, int tid) {
#pragma unroll
  for (int p = 0; p < 8; ++p) {
    int idx = tid + p * 256;
    int row = idx >> 3, c4 = idx & 7;
    r.a[p] = *reinterpret_cast<const float4*>(A + (size_t)(rowA0 + row) * lda + kbase + c4 * 4);
  }
#pragma unroll
  for (int p = 0; p < 4; ++p) {
    int idx = tid + p * 256;
    int row = idx >> 3, c4 = idx & 7;
    r.b[p] = *reinterpret_cast<const float4*>(Bw + (size_t)(rowB0 + row) * ldb + kbase + c4 * 4);
  }
}

// Convert to bf16 and write the staged chunks into an LDS buffer.
__device__ __forceinline__ void stage_store(const StageRegs& r,
                                            unsigned char* sAbuf, unsigned char* sBbuf,
                                            int tid) {
#pragma unroll
  for (int p = 0; p < 8; ++p) {
    int idx = tid + p * 256;
    int row = idx >> 3, c4 = idx & 7;
    *reinterpret_cast<uint2*>(sAbuf + row * LDSROW + c4 * 8) = pack4_bf16(r.a[p]);
  }
#pragma unroll
  for (int p = 0; p < 4; ++p) {
    int idx = tid + p * 256;
    int row = idx >> 3, c4 = idx & 7;
    *reinterpret_cast<uint2*>(sBbuf + row * LDSROW + c4 * 8) = pack4_bf16(r.b[p]);
  }
}

// -------------------------------------------------- kernel 1: gate + dense h

// One wave (32 lanes) per token: with TOP_K=1 the softmax/top-k/renorm collapses
// to argmax-of-logits with weight 1.0.  Compute the active lora's 16-dim h
// (pre-scaled by SCALING) and scatter into a dense h[MTOK][RMOE] fp32 row.
__global__ __launch_bounds__(256) void gate_h_kernel(const float* __restrict__ x,
                                                     const float* __restrict__ gate_w,
                                                     const float* __restrict__ lora_A,
                                                     float* __restrict__ hdense) {
  const int lane = threadIdx.x & 31;
  const int wid  = threadIdx.x >> 5;
  const int m    = blockIdx.x * 8 + wid;
  const float* xr = x + (size_t)m * D_IN;

  // argmax over 29 logits
  float best = -3.0e38f;
  int   bi   = 0;
#pragma unroll 1
  for (int e = 0; e < N_EXP; ++e) {
    const float* gr = gate_w + (size_t)e * D_IN;
    float acc = 0.0f;
    for (int d = lane * 4; d < D_IN; d += 32 * 4) {
      float4 xv = *reinterpret_cast<const float4*>(xr + d);
      float4 gv = *reinterpret_cast<const float4*>(gr + d);
      acc += xv.x * gv.x + xv.y * gv.y + xv.z * gv.z + xv.w * gv.w;
    }
#pragma unroll
    for (int o = 16; o > 0; o >>= 1) acc += __shfl_xor(acc, o, 32);
    if (acc > best) { best = acc; bi = e; }
  }

  // h for the active lora (lane r keeps h[r]); bi==28 means base-only (no delta)
  float myh = 0.0f;
  if (bi < N_LORAS) {
#pragma unroll 1
    for (int r = 0; r < RANK; ++r) {
      const float* ar = lora_A + (size_t)(bi * RANK + r) * D_IN;
      float acc = 0.0f;
      for (int d = lane * 4; d < D_IN; d += 32 * 4) {
        float4 xv = *reinterpret_cast<const float4*>(xr + d);
        float4 av = *reinterpret_cast<const float4*>(ar + d);
        acc += xv.x * av.x + xv.y * av.y + xv.z * av.z + xv.w * av.w;
      }
#pragma unroll
      for (int o = 16; o > 0; o >>= 1) acc += __shfl_xor(acc, o, 32);
      if (lane == r) myh = SCALING * acc;
    }
  }

  // dense scatter: zeros outside the active 16-block, then the block itself
  float* hrow = hdense + (size_t)m * RMOE;
  const int base = (bi < N_LORAS) ? bi * RANK : -1000;
  for (int i = lane; i < RMOE; i += 32) {
    if (i < base || i >= base + RANK) hrow[i] = 0.0f;
  }
  if (bi < N_LORAS && lane < RANK) hrow[base + lane] = myh;
}

// ------------------------------------------------ kernel 2: fused WMMA GEMM

// out[m,n] = sum_k x[m,k]*W[n,k]  +  sum_r h[m,r]*lora_B[n,r]
// Both phases are NT (K-contiguous) -> identical staging/fragment paths.
// Pipeline per K-step: issue global loads (s+1) -> 16 WMMAs (s) -> convert +
// ds_store (s+1) -> barrier.  Load latency hides under the XDL block.
__global__ __launch_bounds__(256) void moe_gemm_kernel(const float* __restrict__ x,
                                                       const float* __restrict__ W,
                                                       const float* __restrict__ loraB,
                                                       const float* __restrict__ hdense,
                                                       float* __restrict__ out) {
  __shared__ __align__(16) unsigned char sA[2][BM * LDSROW];   // 2 x 20 KB
  __shared__ __align__(16) unsigned char sB[2][BN * LDSROW];   // 2 x 10 KB

  const int tid   = threadIdx.x;
  const int lane  = tid & 31;
  const int wid   = tid >> 5;
  const int waveM = wid >> 1;      // 0..3 : 64-row slab
  const int waveN = wid & 1;       // 0..1 : 64-col slab
  const int hsel  = lane >> 4;     // lane half
  const int lrow  = lane & 15;
  const int bm    = blockIdx.y * BM;
  const int bn    = blockIdx.x * BN;

  v8f acc[4][4];
  const v8f vzero = {0.f, 0.f, 0.f, 0.f, 0.f, 0.f, 0.f, 0.f};
#pragma unroll
  for (int mt = 0; mt < 4; ++mt)
#pragma unroll
    for (int nt = 0; nt < 4; ++nt) acc[mt][nt] = vzero;

  auto run_phase = [&](const float* A, int lda, const float* Bw, int ldb, int ksteps) {
    StageRegs regs;
    stage_load(regs, A, lda, Bw, ldb, bm, bn, 0, tid);
    stage_store(regs, sA[0], sB[0], tid);
    __syncthreads();
#pragma unroll 1
    for (int s = 0; s < ksteps; ++s) {
      const int cur = s & 1;
      const bool have_next = (s + 1 < ksteps);

      // 1) issue next tile's global loads early; do NOT consume yet
      if (have_next)
        stage_load(regs, A, lda, Bw, ldb, bm, bn, (s + 1) * KB, tid);

      if (s + 2 < ksteps) {          // hint next-next K tile toward the caches
        __builtin_prefetch(A + (size_t)(bm + tid) * lda + (s + 2) * KB, 0, 0);
        if (tid < BN)
          __builtin_prefetch(Bw + (size_t)(bn + tid) * ldb + (s + 2) * KB, 0, 0);
      }

      // 2) matrix work for the current tile (hides the load latency)
      v16bf aF[4], bF[4];
#pragma unroll
      for (int mt = 0; mt < 4; ++mt)
        aF[mt] = load_frag_a(sA[cur], waveM * 64 + mt * 16 + lrow, hsel);
#pragma unroll
      for (int nt = 0; nt < 4; ++nt)
        bF[nt] = load_frag_b(sB[cur], waveN * 64 + nt * 16 + lrow, hsel);

#pragma unroll
      for (int mt = 0; mt < 4; ++mt)
#pragma unroll
        for (int nt = 0; nt < 4; ++nt)
          acc[mt][nt] = __builtin_amdgcn_wmma_f32_16x16x32_bf16(
              false, aF[mt], false, bF[nt], (short)0, acc[mt][nt], false, false);

      // 3) now consume the staged registers into the other LDS buffer
      if (have_next)
        stage_store(regs, sA[cur ^ 1], sB[cur ^ 1], tid);
      __syncthreads();
    }
  };

  run_phase(x,      D_IN, W,     D_IN, D_IN / KB);   // base GEMM, K=4096
  run_phase(hdense, RMOE, loraB, RMOE, RMOE / KB);   // lora delta, K=448

  // Epilogue: C layout (ISA 7.12.2): lane -> N = lrow, vgpr g -> M = g + 8*hsel
#pragma unroll
  for (int mt = 0; mt < 4; ++mt)
#pragma unroll
    for (int nt = 0; nt < 4; ++nt)
#pragma unroll
      for (int g = 0; g < 8; ++g) {
        int row = bm + waveM * 64 + mt * 16 + g + 8 * hsel;
        int col = bn + waveN * 64 + nt * 16 + lrow;
        out[(size_t)row * D_OUT + col] = acc[mt][nt][g];
      }
}

// ------------------------------------------------------------------ launch --

extern "C" void kernel_launch(void* const* d_in, const int* in_sizes, int n_in,
                              void* d_out, int out_size, void* d_ws, size_t ws_size,
                              hipStream_t stream) {
  const float* x      = (const float*)d_in[0];   // (4,2048,4096)
  const float* baseW  = (const float*)d_in[1];   // (4096,4096)
  const float* gateW  = (const float*)d_in[2];   // (29,4096)
  const float* loraA  = (const float*)d_in[3];   // (448,4096)
  const float* loraB  = (const float*)d_in[4];   // (4096,448)
  float*       out    = (float*)d_out;           // (4,2048,4096)
  float*       hdense = (float*)d_ws;            // MTOK*RMOE f32 = 14.7 MB scratch

  // 1) gate argmax + dense scaled h  (8 tokens per 256-thread block)
  gate_h_kernel<<<MTOK / 8, 256, 0, stream>>>(x, gateW, loraA, hdense);

  // 2) fused base + lora GEMM
  dim3 grid(D_OUT / BN, MTOK / BM);              // (32, 32)
  moe_gemm_kernel<<<grid, 256, 0, stream>>>(x, baseW, loraB, hdense, out);
}
